// VqvaeMnistGradTc_58188216926446
// MI455X (gfx1250) — matrix operand order repaired and checked by reading, not compiled, not run on hardware
//
#include <hip/hip_runtime.h>
#include <hip/hip_bf16.h>

typedef __attribute__((ext_vector_type(16))) _Float16 v16h;
typedef __attribute__((ext_vector_type(8)))  float    v8f;

#define N_ROWS 4096
#define DIM    412
#define DHID   256
#define DE_    160
#define DL_    16
#define KCB    1024
#define NFLAT  (N_ROWS * DL_)     // 65536
#define ZWIDE  (DE_ * DL_)        // 2560

__device__ __forceinline__ float gelu_f(float x) {
  const float c = 0.7978845608028654f;
  return 0.5f * x * (1.0f + tanhf(c * (x + 0.044715f * x * x * x)));
}

// -------------------------------------------------------------------------
// t_emb = t_emb_start*(1-t/MAXT) + t_emb_end*(t/MAXT), shape (N, 256)
// -------------------------------------------------------------------------
__global__ __launch_bounds__(256) void temb_kernel(
    const int* __restrict__ t, const float* __restrict__ s,
    const float* __restrict__ e, float* __restrict__ out) {
  int i = blockIdx.x * blockDim.x + threadIdx.x;
  if (i >= N_ROWS * DHID) return;
  int n = i >> 8, j = i & 255;
  float tt = (float)t[n] * (1.0f / 10000.0f);
  out[i] = s[j] * (1.0f - tt) + e[j] * tt;
}

// -------------------------------------------------------------------------
// codebook row norms
// -------------------------------------------------------------------------
__global__ __launch_bounds__(256) void cb_norm_kernel(
    const float* __restrict__ cb, float* __restrict__ cnorm) {
  int k = blockIdx.x * blockDim.x + threadIdx.x;
  if (k >= KCB) return;
  float s = 0.f;
  for (int e = 0; e < DE_; e++) { float v = cb[k * DE_ + e]; s += v * v; }
  cnorm[k] = s;
}

// -------------------------------------------------------------------------
// WMMA GEMM:  Y[4096 x M] = (GELU? gelu(X) : X)[4096 x K] @ W[K x M]
//                             + bias[M] + (HASC ? Cin : 0)
// Block: 256 threads = 8 waves; block tile 64 rows x 64 cols.
// Wave (wr,wc): wr in 0..1 selects 32-row group (two 16-row WMMA tiles,
// shared B fragment), wc in 0..3 selects 16-col group. 2 WMMAs per wave
// per 32-K step. LDS staging; B staged transposed so each fragment is two
// aligned 16B LDS vector reads per lane (per ISA fragment layouts).
// -------------------------------------------------------------------------
template <int GELU, int HASC>
__global__ __launch_bounds__(256) void gemm_wmma(
    const float* __restrict__ X, const float* __restrict__ W,
    const float* __restrict__ bias, const float* __restrict__ Cin,
    float* __restrict__ Y, int Kdim, int Mdim, int ldY, int ldC) {
  __shared__ __align__(16) _Float16 lA[64 * 32];   // [row][k]
  __shared__ __align__(16) _Float16 lB[64 * 32];   // [col][k] (transposed)

  const int tid  = threadIdx.x;
  const int lane = tid & 31;
  const int wave = tid >> 5;        // 0..7
  const int wr   = wave & 1;        // 32-row group
  const int wc   = wave >> 1;       // 16-col group (0..3)
  const int rowbase = blockIdx.y * 64;
  const int colbase = blockIdx.x * 64;
  const bool interior = (colbase + 64 <= Mdim);

  const int l15 = lane & 15;
  const int h16 = lane >> 4;        // 0/1
  const int ak0 = h16 * 8;          // A frag: K {0..7,16..23} / {8..15,24..31}
  const int bk0 = h16 * 8;          // B frag: K {0..7,16..23} / {8..15,24..31}
  const int arow0 = wr * 32 + l15;  // first 16-row tile
  const int arow1 = arow0 + 16;     // second 16-row tile
  const int bcol  = wc * 16 + l15;

  v8f acc0 = {}, acc1 = {};

  const int kfull = Kdim & ~31;
  int kb = 0;
  for (; kb < kfull; kb += 32) {
    __syncthreads();
    // stage A tile (64 rows x 32 k), no K guard (full chunk)
    for (int i = tid; i < 64 * 32; i += 256) {
      int r = i >> 5, k = i & 31;
      float v = X[(rowbase + r) * Kdim + (kb + k)];
      if (GELU) v = gelu_f(v);
      lA[i] = (_Float16)v;
    }
    // stage B tile transposed (64 cols x 32 k)
    if (interior) {
      for (int i = tid; i < 64 * 32; i += 256) {
        int cc = i & 63, k = i >> 6;
        lB[cc * 32 + k] = (_Float16)W[(kb + k) * Mdim + (colbase + cc)];
      }
    } else {
      for (int i = tid; i < 64 * 32; i += 256) {
        int cc = i & 63, k = i >> 6;
        int gc = colbase + cc;
        float v = (gc < Mdim) ? W[(kb + k) * Mdim + gc] : 0.f;
        lB[cc * 32 + k] = (_Float16)v;
      }
    }
    __syncthreads();

    union { uint4 u[2]; v16h h; } ua0, ua1, ub;
    ub.u[0]  = *(const uint4*)&lB[bcol * 32 + bk0];
    ub.u[1]  = *(const uint4*)&lB[bcol * 32 + bk0 + 16];
    ua0.u[0] = *(const uint4*)&lA[arow0 * 32 + ak0];
    ua0.u[1] = *(const uint4*)&lA[arow0 * 32 + ak0 + 16];
    ua1.u[0] = *(const uint4*)&lA[arow1 * 32 + ak0];
    ua1.u[1] = *(const uint4*)&lA[arow1 * 32 + ak0 + 16];
    acc0 = __builtin_amdgcn_wmma_f32_16x16x32_f16(
        false, ua0.h, false, ub.h, (short)0, acc0, false, false);
    acc1 = __builtin_amdgcn_wmma_f32_16x16x32_f16(
        false, ua1.h, false, ub.h, (short)0, acc1, false, false);
  }

  if (kfull < Kdim) {                       // K remainder chunk (only K=412)
    __syncthreads();
    for (int i = tid; i < 64 * 32; i += 256) {
      int r = i >> 5, k = i & 31;
      int gk = kb + k;
      float v = 0.f;
      if (gk < Kdim) {
        v = X[(rowbase + r) * Kdim + gk];
        if (GELU) v = gelu_f(v);
      }
      lA[i] = (_Float16)v;
    }
    for (int i = tid; i < 64 * 32; i += 256) {
      int cc = i & 63, k = i >> 6;
      int gk = kb + k, gc = colbase + cc;
      float v = (gk < Kdim && gc < Mdim) ? W[gk * Mdim + gc] : 0.f;
      lB[cc * 32 + k] = (_Float16)v;
    }
    __syncthreads();
    union { uint4 u[2]; v16h h; } ua0, ua1, ub;
    ub.u[0]  = *(const uint4*)&lB[bcol * 32 + bk0];
    ub.u[1]  = *(const uint4*)&lB[bcol * 32 + bk0 + 16];
    ua0.u[0] = *(const uint4*)&lA[arow0 * 32 + ak0];
    ua0.u[1] = *(const uint4*)&lA[arow0 * 32 + ak0 + 16];
    ua1.u[0] = *(const uint4*)&lA[arow1 * 32 + ak0];
    ua1.u[1] = *(const uint4*)&lA[arow1 * 32 + ak0 + 16];
    acc0 = __builtin_amdgcn_wmma_f32_16x16x32_f16(
        false, ua0.h, false, ub.h, (short)0, acc0, false, false);
    acc1 = __builtin_amdgcn_wmma_f32_16x16x32_f16(
        false, ua1.h, false, ub.h, (short)0, acc1, false, false);
  }

  // epilogue: C/D layout -> VGPR v: row v (+8 for lanes 16..31), col lane&15
  int gcol = colbase + wc * 16 + l15;
  if (gcol < Mdim) {
    float b = bias[gcol];
#pragma unroll
    for (int v = 0; v < 8; v++) {
      int grow = rowbase + wr * 32 + v + h16 * 8;
      float o = acc0[v] + b;
      if (HASC) o += Cin[grow * ldC + gcol];
      Y[grow * ldY + gcol] = o;
    }
#pragma unroll
    for (int v = 0; v < 8; v++) {
      int grow = rowbase + wr * 32 + 16 + v + h16 * 8;
      float o = acc1[v] + b;
      if (HASC) o += Cin[grow * ldC + gcol];
      Y[grow * ldY + gcol] = o;
    }
  }
}

// -------------------------------------------------------------------------
// VQ argmin: flat (65536 x 160, strided view of z_e) vs codebook (1024 x 160)
// One wave per 16 flat rows. score = ||c||^2 - 2*f.c  (||f||^2 const/row).
// -------------------------------------------------------------------------
__global__ __launch_bounds__(256) void vq_argmin_kernel(
    const float* __restrict__ z_e, const float* __restrict__ cb,
    const float* __restrict__ cnorm, int* __restrict__ idx_i,
    float* __restrict__ idx_f) {
  const int tid  = threadIdx.x;
  const int lane = tid & 31;
  const int wave = tid >> 5;
  const int rtile = blockIdx.x * 8 + wave;   // 0..4095
  const int rbase = rtile * 16;
  const int l15 = lane & 15, h16 = lane >> 4;

  // this lane's flat row for the A fragment
  const int r = rbase + l15;
  const int n = r >> 4, l = r & 15;
  const float* zrow = z_e + n * ZWIDE + l;   // stride DL_=16 floats per e

  // preload A fragments for all 5 K-chunks (K=160 = 5*32)
  v16h a[5];
#pragma unroll
  for (int kk = 0; kk < 5; kk++) {
    int k0 = kk * 32 + h16 * 16;
#pragma unroll
    for (int j = 0; j < 8; j++) {
      a[kk][j]     = (_Float16)zrow[(k0 + j) * DL_];
      a[kk][j + 8] = (_Float16)zrow[(k0 + 8 + j) * DL_];
    }
  }

  float mval[8];
  int   midx[8];
#pragma unroll
  for (int v = 0; v < 8; v++) { mval[v] = 3.4e38f; midx[v] = 0; }

  for (int cc = 0; cc < KCB / 16; cc++) {
    const int code = cc * 16 + l15;                 // this lane's column/code
    const float* crow = cb + code * DE_;
    v8f acc = {};
#pragma unroll
    for (int kk = 0; kk < 5; kk++) {
      int k0 = kk * 32 + h16 * 16;                  // 16B-aligned, contiguous
      const float4* p = (const float4*)(crow + k0);
      float4 q0 = p[0], q1 = p[1], q2 = p[2], q3 = p[3];
      v16h b;
      b[0]  = (_Float16)q0.x; b[1]  = (_Float16)q0.y;
      b[2]  = (_Float16)q0.z; b[3]  = (_Float16)q0.w;
      b[4]  = (_Float16)q1.x; b[5]  = (_Float16)q1.y;
      b[6]  = (_Float16)q1.z; b[7]  = (_Float16)q1.w;
      b[8]  = (_Float16)q2.x; b[9]  = (_Float16)q2.y;
      b[10] = (_Float16)q2.z; b[11] = (_Float16)q2.w;
      b[12] = (_Float16)q3.x; b[13] = (_Float16)q3.y;
      b[14] = (_Float16)q3.z; b[15] = (_Float16)q3.w;
      acc = __builtin_amdgcn_wmma_f32_16x16x32_f16(
          false, a[kk], false, b, (short)0, acc, false, false);
    }
    float cn = cnorm[code];
#pragma unroll
    for (int v = 0; v < 8; v++) {
      float s = cn - 2.0f * acc[v];
      if (s < mval[v]) { mval[v] = s; midx[v] = code; }  // strict < : first-min
    }
  }

  // reduce over the 16 columns held by each 16-lane half (rows stay in-half)
#pragma unroll
  for (int v = 0; v < 8; v++) {
    float bv = mval[v];
    int   bi = midx[v];
    for (int off = 1; off < 16; off <<= 1) {
      float ov = __shfl_xor(bv, off, 32);
      int   oi = __shfl_xor(bi, off, 32);
      if (ov < bv || (ov == bv && oi < bi)) { bv = ov; bi = oi; }
    }
    if (l15 == 0) {
      int row = rbase + v + h16 * 8;
      idx_i[row] = bi;
      idx_f[row] = (float)bi;
    }
  }
}

// -------------------------------------------------------------------------
// z_q[n, e*16 + l] = codebook[idx[n*16+l], e]
// -------------------------------------------------------------------------
__global__ __launch_bounds__(256) void vq_gather_kernel(
    const int* __restrict__ idx_i, const float* __restrict__ cb,
    float* __restrict__ zq) {
  int i = blockIdx.x * blockDim.x + threadIdx.x;
  if (i >= NFLAT * DE_) return;
  int r = i / DE_, e = i - r * DE_;
  int n = r >> 4, l = r & 15;
  zq[n * ZWIDE + e * DL_ + l] = cb[idx_i[r] * DE_ + e];
}

// -------------------------------------------------------------------------
// host side
// -------------------------------------------------------------------------
static inline void launch_gemm(const float* X, const float* W, const float* b,
                               const float* Cin, float* Y, int K, int M,
                               int ldY, int ldC, int gelu_in, hipStream_t s) {
  dim3 grid((M + 63) / 64, N_ROWS / 64);
  if (gelu_in) {
    if (Cin) gemm_wmma<1, 1><<<grid, 256, 0, s>>>(X, W, b, Cin, Y, K, M, ldY, ldC);
    else     gemm_wmma<1, 0><<<grid, 256, 0, s>>>(X, W, b, Cin, Y, K, M, ldY, ldC);
  } else {
    if (Cin) gemm_wmma<0, 1><<<grid, 256, 0, s>>>(X, W, b, Cin, Y, K, M, ldY, ldC);
    else     gemm_wmma<0, 0><<<grid, 256, 0, s>>>(X, W, b, Cin, Y, K, M, ldY, ldC);
  }
}

extern "C" void kernel_launch(void* const* d_in, const int* in_sizes, int n_in,
                              void* d_out, int out_size, void* d_ws, size_t ws_size,
                              hipStream_t stream) {
  // ---- inputs (setup_inputs dict insertion order, leaves w-then-b) ----
  const float* x     = (const float*)d_in[0];   // (4096,412)
  const int*   t     = (const int*)  d_in[1];   // (4096,)
  const float* c     = (const float*)d_in[2];   // (4096,412)
  const float* tes   = (const float*)d_in[3];   // (1,256)
  const float* tee   = (const float*)d_in[4];   // (1,256)
  // si_params: fc.w fc.b fc_t.w fc_t.b res{l1,l2,l3}{w,b}  -> d_in[5..14]
  // enc_params: fc_in, fc_cond, res1(l1..l3), fc1, res2(l1..l3), fc2 -> [15..34]
  // dec_params: same layout -> [35..54]
  const float* cb    = (const float*)d_in[55];  // (1024,160)
  auto F = [&](int i) { return (const float*)d_in[i]; };

  // ---- output regions (concatenated, return order) ----
  float* out = (float*)d_out;
  float* o_xrec = out;                                  // 4096*412
  float* o_zq   = o_xrec + (size_t)N_ROWS * DIM;        // 4096*2560
  float* o_cb   = o_zq   + (size_t)N_ROWS * ZWIDE;      // 1024*160
  float* o_ze   = o_cb   + (size_t)KCB * DE_;           // 4096*2560
  float* o_idx  = o_ze   + (size_t)N_ROWS * ZWIDE;      // 65536

  // ---- workspace ----
  float* ws = (float*)d_ws;
  size_t o = 0;
  float* temb   = ws + o; o += (size_t)N_ROWS * DHID;   // (N,256)
  float* h      = ws + o; o += (size_t)N_ROWS * DIM;    // SINet trunk (N,412)
  float* cond   = ws + o; o += (size_t)N_ROWS * DIM;    // (N,412)
  float* xc     = ws + o; o += (size_t)N_ROWS * DHID;   // concat buf (N,256)
  float* xc2    = ws + o; o += (size_t)N_ROWS * DHID;   // (N,256)
  float* r256   = ws + o; o += (size_t)N_ROWS * DHID;   // resblock out (N,256)
  float* tA     = ws + o; o += (size_t)N_ROWS * DE_;    // (N,160)
  float* tB     = ws + o; o += (size_t)N_ROWS * DE_;    // (N,160)
  float* cnorm  = ws + o; o += KCB;
  int*   idxi   = (int*)(ws + o); o += NFLAT;
  float* tmp412 = ws + o; o += (size_t)N_ROWS * DIM;    // (N,412)

  // 1) t_emb
  temb_kernel<<<(N_ROWS * DHID + 255) / 256, 256, 0, stream>>>(t, tes, tee, temb);

  // 2) SINet: h = c@fc + fc_t(temb);  cond = resblock(h)
  launch_gemm(c,    F(5), F(6),  nullptr, tmp412, DIM,  DIM, DIM, 0,   0, stream);
  launch_gemm(temb, F(7), F(8),  tmp412,  h,      DHID, DIM, DIM, DIM, 0, stream);
  launch_gemm(h,  F(9),  F(10), nullptr, tA,   DIM, DE_, DE_, 0,   1, stream);
  launch_gemm(tA, F(11), F(12), nullptr, tB,   DE_, DE_, DE_, 0,   1, stream);
  launch_gemm(tB, F(13), F(14), h,       cond, DE_, DIM, DIM, DIM, 1, stream);

  // 3) Encoder
  launch_gemm(x,    F(15), F(16), temb,       xc,       DIM, 128, DHID, DHID, 0, stream);
  launch_gemm(cond, F(17), F(18), temb + 128, xc + 128, DIM, 128, DHID, DHID, 0, stream);
  launch_gemm(xc, F(19), F(20), nullptr, tA,   DHID, DE_,  DE_,  0,    1, stream);
  launch_gemm(tA, F(21), F(22), nullptr, tB,   DE_,  DE_,  DE_,  0,    1, stream);
  launch_gemm(tB, F(23), F(24), xc,      r256, DE_,  DHID, DHID, DHID, 1, stream);
  launch_gemm(r256, F(25), F(26), nullptr, xc2, DHID, DHID, DHID, 0,   1, stream);
  launch_gemm(xc2, F(27), F(28), nullptr, tA,   DHID, DE_,  DE_,  0,    1, stream);
  launch_gemm(tA,  F(29), F(30), nullptr, tB,   DE_,  DE_,  DE_,  0,    1, stream);
  launch_gemm(tB,  F(31), F(32), xc2,     r256, DE_,  DHID, DHID, DHID, 1, stream);
  launch_gemm(r256, F(33), F(34), nullptr, o_ze, DHID, ZWIDE, ZWIDE, 0, 1, stream); // z_e

  // 4) VQ
  cb_norm_kernel<<<(KCB + 255) / 256, 256, 0, stream>>>(cb, cnorm);
  vq_argmin_kernel<<<NFLAT / 16 / 8, 256, 0, stream>>>(o_ze, cb, cnorm, idxi, o_idx);
  vq_gather_kernel<<<((size_t)NFLAT * DE_ + 255) / 256, 256, 0, stream>>>(idxi, cb, o_zq);
  hipMemcpyAsync(o_cb, cb, (size_t)KCB * DE_ * sizeof(float),
                 hipMemcpyDeviceToDevice, stream);

  // 5) Decoder (input = z_q)
  launch_gemm(o_zq, F(35), F(36), temb,       xc,       ZWIDE, 128, DHID, DHID, 0, stream);
  launch_gemm(cond, F(37), F(38), temb + 128, xc + 128, DIM,   128, DHID, DHID, 0, stream);
  launch_gemm(xc, F(39), F(40), nullptr, tA,   DHID, DE_,  DE_,  0,    1, stream);
  launch_gemm(tA, F(41), F(42), nullptr, tB,   DE_,  DE_,  DE_,  0,    1, stream);
  launch_gemm(tB, F(43), F(44), xc,      r256, DE_,  DHID, DHID, DHID, 1, stream);
  launch_gemm(r256, F(45), F(46), nullptr, xc2, DHID, DHID, DHID, 0,   1, stream);
  launch_gemm(xc2, F(47), F(48), nullptr, tA,   DHID, DE_,  DE_,  0,    1, stream);
  launch_gemm(tA,  F(49), F(50), nullptr, tB,   DE_,  DE_,  DE_,  0,    1, stream);
  launch_gemm(tB,  F(51), F(52), xc2,     r256, DE_,  DHID, DHID, DHID, 1, stream);
  launch_gemm(r256, F(53), F(54), nullptr, o_xrec, DHID, DIM, DIM, 0,  1, stream); // x_rec

  (void)in_sizes; (void)n_in; (void)out_size; (void)ws_size;
}
